// VectorQuantizer_69415261438322
// MI455X (gfx1250) — compile-verified, matrix operand order checked by reading, not run on hardware
//
#include <hip/hip_runtime.h>

// CDNA5 / gfx1250 VQ-VAE vector quantizer.
// Distances via v_wmma_f32_16x16x32_bf16 with register-resident A fragments and
// two independent accumulator chains. The bf16 B-fragment codebook (+ norms) is
// precomputed once into d_ws (L2-resident), so the sweep loop stages only a
// 32 KB coalesced copy per 32 codes. Argmin via lane butterfly; f32 gather z_q.

typedef __attribute__((ext_vector_type(16))) __bf16 v16bf;
typedef __attribute__((ext_vector_type(8)))  float  v8f;

#define DDIM    512
#define KCODES  4096
#define NROWS   32768                 // 8 * 4096
#define RPB     128                   // rows per block
#define WAVES   8                     // 256 threads, wave32
#define NT      2                     // code tiles processed per sweep (N = 32)
#define TILE    16                    // codes per WMMA tile
#define SWEEP   (NT * TILE)           // 32 codes per sweep
#define NSWEEPS (KCODES / SWEEP)      // 128
#define KCHUNKS (DDIM / 32)           // 16 WMMA k-steps
#define CBPAD   513                   // padded row stride (bank-conflict free)

// d_ws layout (precompute path)
#define FRAG_USHORTS  ((size_t)KCODES * DDIM)            // 2M bf16 = 4 MB
#define FRAG_BYTES    (FRAG_USHORTS * 2)
#define NORM_BYTES    ((size_t)KCODES * 4)               // 16 KB
#define WS_NEED       (FRAG_BYTES + NORM_BYTES + 4)

__device__ __forceinline__ unsigned short f32_to_bf16(float f) {
  unsigned int u = __float_as_uint(f);
  u += 0x7FFFu + ((u >> 16) & 1u);    // round-to-nearest-even
  return (unsigned short)(u >> 16);
}

// ---------------------------------------------------------------------------
// Precompute: bf16 B fragments (ISA 7.12.2 16-bit B layout) + f32 norms.
// One block per 16-code tile. Fragment slot s = kc*32 + lane; 16 bf16 each.
// ---------------------------------------------------------------------------
__global__ __launch_bounds__(256, 1)
void vq_precompute(const float* __restrict__ cb,
                   unsigned short* __restrict__ fragG,
                   float* __restrict__ normG) {
  __shared__ float cbF32[TILE * CBPAD];
  __shared__ float normPart[TILE * 4];

  const int tid      = threadIdx.x;
  const int tile     = blockIdx.x;
  const int codeBase = tile * TILE;

  for (int i = 0; i < (TILE * DDIM) / 256; ++i) {
    int idx = tid + i * 256;
    int n = idx >> 9, col = idx & 511;
    cbF32[n * CBPAD + col] = cb[(size_t)(codeBase + n) * DDIM + col];
  }
  __syncthreads();

  if (tid < TILE * 4) {
    int code = tid >> 2, part = tid & 3;
    const float* rp = &cbF32[code * CBPAD + part * 128];
    float s = 0.f;
    for (int c = 0; c < 128; ++c) { float v = rp[c]; s += v * v; }
    normPart[tid] = s;
  }
  __syncthreads();
  if (tid < TILE)
    normG[codeBase + tid] = normPart[tid * 4] + normPart[tid * 4 + 1] +
                            normPart[tid * 4 + 2] + normPart[tid * 4 + 3];

  // 512 slots per tile; each thread packs 2 slots (32 B each) via uint4 stores
  for (int s = tid * 2; s < tid * 2 + 2; ++s) {
    int kc = s >> 5, ln = s & 31;
    int n  = ln & 15, g = ln >> 4;
    const float* src = &cbF32[n * CBPAD + kc * 32];
    unsigned int w[8];
#pragma unroll
    for (int j = 0; j < 8; ++j) {
      int e0 = 2 * j, e1 = 2 * j + 1;
      int K0 = (e0 < 8) ? (g * 8 + e0) : (16 + g * 8 + (e0 - 8));
      int K1 = (e1 < 8) ? (g * 8 + e1) : (16 + g * 8 + (e1 - 8));
      w[j] = (unsigned int)f32_to_bf16(src[K0]) |
             ((unsigned int)f32_to_bf16(src[K1]) << 16);
    }
    uint4* dst = (uint4*)&fragG[((size_t)tile * 512 + s) * 16];
    dst[0] = make_uint4(w[0], w[1], w[2], w[3]);
    dst[1] = make_uint4(w[4], w[5], w[6], w[7]);
  }
}

// ---------------------------------------------------------------------------
// Main kernel (precomputed-fragment path)
// ---------------------------------------------------------------------------
__global__ __launch_bounds__(256, 1)
void vq_main_pre(const float* __restrict__ z, const float* __restrict__ cb,
                 const unsigned short* __restrict__ fragG,
                 const float* __restrict__ normG,
                 float* __restrict__ zq_out, int* __restrict__ idx_out,
                 float* __restrict__ loss_acc) {
  __shared__ __align__(32) unsigned short aLds[WAVES * KCHUNKS * 32 * 16]; // 128KB
  __shared__ __align__(32) unsigned short bLds[NT * KCHUNKS * 32 * 16];    // 32KB
  __shared__ int   idxLds[RPB];
  __shared__ float lossW[WAVES];

  const int tid     = threadIdx.x;
  const int lane    = tid & 31;
  const int wave    = tid >> 5;
  const int rowBase = blockIdx.x * RPB;

  // ---- Phase 0: z tile -> bf16 A fragments in LDS, then hoist to registers
  for (int i = 0; i < (RPB * DDIM) / 256; ++i) {
    int idx = tid + i * 256;
    int row = idx >> 9;
    int col = idx & 511;
    float f = z[(size_t)(rowBase + row) * DDIM + col];
    int w    = row >> 4, m = row & 15;
    int kc   = col >> 5, kloc = col & 31;
    int g    = (kloc >> 3) & 1;
    int e    = (kloc & 7) + ((kloc >> 4) << 3);
    int ln   = m + (g << 4);
    aLds[(((w * KCHUNKS + kc) * 32) + ln) * 16 + e] = f32_to_bf16(f);
  }
  __syncthreads();

  v16bf a_reg[KCHUNKS];                  // 128 VGPRs, sweep-invariant
#pragma unroll
  for (int kc = 0; kc < KCHUNKS; ++kc)
    a_reg[kc] = *reinterpret_cast<const v16bf*>(
        &aLds[((wave * KCHUNKS + kc) * 32 + lane) * 16]);

  float best[8];
  int   bidx[8];
#pragma unroll
  for (int r = 0; r < 8; ++r) { best[r] = 3.0e38f; bidx[r] = 0; }

  const unsigned short* bBase = &bLds[lane * 16];   // + imm byte offsets below

  // ---- Phase 1: sweep the codebook, 32 codes (2 fragment tiles) at a time
  for (int t = 0; t < NSWEEPS; ++t) {
    const int codeBase = t * SWEEP;
    __syncthreads();   // previous sweep fully consumed

    // 32 KB coalesced copy: ready-made fragments global -> LDS (8 uint4/thread)
    {
      const uint4* src = (const uint4*)&fragG[(size_t)t * (NT * 512) * 16];
      uint4*       dst = (uint4*)bLds;
#pragma unroll
      for (int i = 0; i < 8; ++i) dst[tid + i * 256] = src[tid + i * 256];
    }
    if (t + 1 < NSWEEPS) {
      const char* nxt = (const char*)&fragG[(size_t)(t + 1) * (NT * 512) * 16];
      __builtin_prefetch(nxt + tid * 128, 0, 1);   // global_prefetch_b8
    }
    __syncthreads();

    // two independent 16x16 accumulator chains over D=512 (32 WMMAs)
    v8f acc0 = {0.f, 0.f, 0.f, 0.f, 0.f, 0.f, 0.f, 0.f};
    v8f acc1 = {0.f, 0.f, 0.f, 0.f, 0.f, 0.f, 0.f, 0.f};
#pragma unroll
    for (int kc = 0; kc < KCHUNKS; ++kc) {
      v16bf b0 = *reinterpret_cast<const v16bf*>(bBase + kc * (32 * 16));
      v16bf b1 = *reinterpret_cast<const v16bf*>(bBase + (KCHUNKS + kc) * (32 * 16));
      acc0 = __builtin_amdgcn_wmma_f32_16x16x32_bf16(
          false, a_reg[kc], false, b0, (short)0, acc0, false, false);
      acc1 = __builtin_amdgcn_wmma_f32_16x16x32_bf16(
          false, a_reg[kc], false, b1, (short)0, acc1, false, false);
    }

    // dist = ||c||^2 - 2*dot ; running per-lane argmin (N = lane&15)
    int   code0 = codeBase + (lane & 15);
    int   code1 = code0 + TILE;
    float cn0   = normG[code0];
    float cn1   = normG[code1];
#pragma unroll
    for (int r = 0; r < 8; ++r) {
      float d0 = cn0 - 2.0f * acc0[r];
      float d1 = cn1 - 2.0f * acc1[r];
      if (d0 < best[r]) { best[r] = d0; bidx[r] = code0; }
      if (d1 < best[r]) { best[r] = d1; bidx[r] = code1; }
    }
  }

  // ---- Phase 2: reduce argmin across the 16 lanes of each N-group
#pragma unroll
  for (int r = 0; r < 8; ++r) {
    float v  = best[r];
    int   ix = bidx[r];
#pragma unroll
    for (int m = 8; m >= 1; m >>= 1) {
      float ov = __shfl_xor(v, m, 32);
      int   oi = __shfl_xor(ix, m, 32);
      if (ov < v || (ov == v && oi < ix)) { v = ov; ix = oi; }
    }
    int row = wave * 16 + r + ((lane >> 4) << 3);   // M = r + 8*(lane/16)
    if ((lane & 15) == 0) {
      idxLds[row] = ix;
      idx_out[rowBase + row] = ix;
    }
  }
  __syncthreads();

  // ---- Phase 3: gather z_q = codebook[idx] (f32), write out, accumulate loss
  float lsum = 0.f;
  for (int r0 = 0; r0 < RPB; ++r0) {
    int ci   = idxLds[r0];
    int grow = rowBase + r0;
    for (int c = tid; c < DDIM; c += 256) {
      float q  = cb[(size_t)ci * DDIM + c];
      float zz = z[(size_t)grow * DDIM + c];
      float df = q - zz;
      lsum += df * df;
      zq_out[(size_t)grow * DDIM + c] = q;   // z_q_ste == z_q numerically
    }
  }
#pragma unroll
  for (int m = 16; m >= 1; m >>= 1) lsum += __shfl_xor(lsum, m, 32);
  if (lane == 0) lossW[wave] = lsum;
  __syncthreads();
  if (tid == 0) {
    float s = 0.f;
    for (int w = 0; w < WAVES; ++w) s += lossW[w];
    atomicAdd(loss_acc, s);
  }
}

// ---------------------------------------------------------------------------
// Self-contained fallback (round-2 kernel): used only if ws_size < WS_NEED.
// ---------------------------------------------------------------------------
__global__ __launch_bounds__(256, 1)
void vq_main_self(const float* __restrict__ z, const float* __restrict__ cb,
                  float* __restrict__ zq_out, int* __restrict__ idx_out,
                  float* __restrict__ loss_acc) {
  __shared__ __align__(32) unsigned short aLds[WAVES * KCHUNKS * 32 * 16];
  __shared__ __align__(32) unsigned short bLds[NT * KCHUNKS * 32 * 16];
  __shared__ float cbF32[SWEEP * CBPAD];
  __shared__ float cbNorm[SWEEP];
  __shared__ float normPart[SWEEP * 4];
  __shared__ int   idxLds[RPB];
  __shared__ float lossW[WAVES];

  const int tid     = threadIdx.x;
  const int lane    = tid & 31;
  const int wave    = tid >> 5;
  const int rowBase = blockIdx.x * RPB;

  for (int i = 0; i < (RPB * DDIM) / 256; ++i) {
    int idx = tid + i * 256;
    int row = idx >> 9;
    int col = idx & 511;
    float f = z[(size_t)(rowBase + row) * DDIM + col];
    int w    = row >> 4, m = row & 15;
    int kc   = col >> 5, kloc = col & 31;
    int g    = (kloc >> 3) & 1;
    int e    = (kloc & 7) + ((kloc >> 4) << 3);
    int ln   = m + (g << 4);
    aLds[(((w * KCHUNKS + kc) * 32) + ln) * 16 + e] = f32_to_bf16(f);
  }
  __syncthreads();

  v16bf a_reg[KCHUNKS];
#pragma unroll
  for (int kc = 0; kc < KCHUNKS; ++kc)
    a_reg[kc] = *reinterpret_cast<const v16bf*>(
        &aLds[((wave * KCHUNKS + kc) * 32 + lane) * 16]);

  float best[8];
  int   bidx[8];
#pragma unroll
  for (int r = 0; r < 8; ++r) { best[r] = 3.0e38f; bidx[r] = 0; }

  for (int t = 0; t < NSWEEPS; ++t) {
    const int codeBase = t * SWEEP;
    __syncthreads();

    for (int i = 0; i < (SWEEP * DDIM) / 256; ++i) {
      int idx = tid + i * 256;
      int n = idx >> 9, col = idx & 511;
      cbF32[n * CBPAD + col] = cb[(size_t)(codeBase + n) * DDIM + col];
    }
    if (t + 1 < NSWEEPS) {
      const float* nxt = cb + (size_t)(codeBase + SWEEP) * DDIM;
      __builtin_prefetch(nxt + tid * 16, 0, 1);
      __builtin_prefetch(nxt + 4096 + tid * 16, 0, 1);
    }
    __syncthreads();

    if (tid < SWEEP * 4) {
      int code = tid >> 2, part = tid & 3;
      const float* rp = &cbF32[code * CBPAD + part * 128];
      float s = 0.f;
      for (int c = 0; c < 128; ++c) { float v = rp[c]; s += v * v; }
      normPart[tid] = s;
    }
    for (int s = tid * 4; s < tid * 4 + 4; ++s) {
      int half = s >> 9;
      int kc   = (s >> 5) & 15;
      int ln   = s & 31;
      int n    = ln & 15, g = ln >> 4;
      unsigned short* dst = &bLds[s * 16];
      const float*    src = &cbF32[(half * TILE + n) * CBPAD + kc * 32];
#pragma unroll
      for (int e = 0; e < 16; ++e) {
        int K = (e < 8) ? (g * 8 + e) : (16 + g * 8 + (e - 8));
        dst[e] = f32_to_bf16(src[K]);
      }
    }
    __syncthreads();
    if (tid < SWEEP)
      cbNorm[tid] = normPart[tid * 4] + normPart[tid * 4 + 1] +
                    normPart[tid * 4 + 2] + normPart[tid * 4 + 3];
    __syncthreads();

    v8f acc0 = {0.f, 0.f, 0.f, 0.f, 0.f, 0.f, 0.f, 0.f};
    v8f acc1 = {0.f, 0.f, 0.f, 0.f, 0.f, 0.f, 0.f, 0.f};
#pragma unroll
    for (int kc = 0; kc < KCHUNKS; ++kc) {
      v16bf b0 = *reinterpret_cast<const v16bf*>(
          &bLds[((0 * KCHUNKS + kc) * 32 + lane) * 16]);
      v16bf b1 = *reinterpret_cast<const v16bf*>(
          &bLds[((1 * KCHUNKS + kc) * 32 + lane) * 16]);
      acc0 = __builtin_amdgcn_wmma_f32_16x16x32_bf16(
          false, a_reg[kc], false, b0, (short)0, acc0, false, false);
      acc1 = __builtin_amdgcn_wmma_f32_16x16x32_bf16(
          false, a_reg[kc], false, b1, (short)0, acc1, false, false);
    }

    float cn0   = cbNorm[lane & 15];
    float cn1   = cbNorm[TILE + (lane & 15)];
    int   code0 = codeBase + (lane & 15);
    int   code1 = code0 + TILE;
#pragma unroll
    for (int r = 0; r < 8; ++r) {
      float d0 = cn0 - 2.0f * acc0[r];
      float d1 = cn1 - 2.0f * acc1[r];
      if (d0 < best[r]) { best[r] = d0; bidx[r] = code0; }
      if (d1 < best[r]) { best[r] = d1; bidx[r] = code1; }
    }
  }

#pragma unroll
  for (int r = 0; r < 8; ++r) {
    float v  = best[r];
    int   ix = bidx[r];
#pragma unroll
    for (int m = 8; m >= 1; m >>= 1) {
      float ov = __shfl_xor(v, m, 32);
      int   oi = __shfl_xor(ix, m, 32);
      if (ov < v || (ov == v && oi < ix)) { v = ov; ix = oi; }
    }
    int row = wave * 16 + r + ((lane >> 4) << 3);
    if ((lane & 15) == 0) {
      idxLds[row] = ix;
      idx_out[rowBase + row] = ix;
    }
  }
  __syncthreads();

  float lsum = 0.f;
  for (int r0 = 0; r0 < RPB; ++r0) {
    int ci   = idxLds[r0];
    int grow = rowBase + r0;
    for (int c = tid; c < DDIM; c += 256) {
      float q  = cb[(size_t)ci * DDIM + c];
      float zz = z[(size_t)grow * DDIM + c];
      float df = q - zz;
      lsum += df * df;
      zq_out[(size_t)grow * DDIM + c] = q;
    }
  }
#pragma unroll
  for (int m = 16; m >= 1; m >>= 1) lsum += __shfl_xor(lsum, m, 32);
  if (lane == 0) lossW[wave] = lsum;
  __syncthreads();
  if (tid == 0) {
    float s = 0.f;
    for (int w = 0; w < WAVES; ++w) s += lossW[w];
    atomicAdd(loss_acc, s);
  }
}

__global__ void vq_finalize(const float* __restrict__ acc, float* __restrict__ out) {
  // vq_loss = mean((z_q - z)^2) * (1 + BETA), BETA = 0.25
  *out = acc[0] * 1.25f / 16777216.0f;   // numel = 32768 * 512
}

extern "C" void kernel_launch(void* const* d_in, const int* in_sizes, int n_in,
                              void* d_out, int out_size, void* d_ws, size_t ws_size,
                              hipStream_t stream) {
  const float* z  = (const float*)d_in[0];   // (8,4096,512) f32
  const float* cb = (const float*)d_in[1];   // (4096,512)  f32

  float* zq       = (float*)d_out;                       // 16,777,216 elems
  const size_t ZQ = (size_t)NROWS * DDIM;
  float* lossOut  = zq + ZQ;                             // 1 elem
  int*   idxOut   = (int*)(zq + ZQ + 1);                 // 32,768 elems

  if (ws_size >= WS_NEED) {
    unsigned short* fragG = (unsigned short*)d_ws;
    float*          normG = (float*)((char*)d_ws + FRAG_BYTES);
    float*          acc   = (float*)((char*)d_ws + FRAG_BYTES + NORM_BYTES);
    hipMemsetAsync(acc, 0, sizeof(float), stream);
    vq_precompute<<<KCODES / TILE, 256, 0, stream>>>(cb, fragG, normG);
    vq_main_pre<<<NROWS / RPB, 256, 0, stream>>>(z, cb, fragG, normG,
                                                 zq, idxOut, acc);
    vq_finalize<<<1, 1, 0, stream>>>(acc, lossOut);
  } else {
    float* acc = (float*)d_ws;
    hipMemsetAsync(acc, 0, sizeof(float), stream);
    vq_main_self<<<NROWS / RPB, 256, 0, stream>>>(z, cb, zq, idxOut, acc);
    vq_finalize<<<1, 1, 0, stream>>>(acc, lossOut);
  }
}
